// ActualConfig9Model_70789650972771
// MI455X (gfx1250) — matrix operand order repaired and checked by reading, not compiled, not run on hardware
//
#include <hip/hip_runtime.h>
#include <math.h>

#define NN   50000
#define NE   200000
#define NG   256
#define DIN  100
#define KPIN 128
#define HD   384
#define H3   1152
#define HH   192
#define GSTEPS 4
#define BN_EPS 1e-5f

typedef __bf16 bf16;
typedef __attribute__((ext_vector_type(16))) __bf16 v16bf;
typedef __attribute__((ext_vector_type(8)))  __bf16 v8bf;
typedef __attribute__((ext_vector_type(8)))  float  v8f;

static __device__ __forceinline__ bf16 to_bf16(float f) {
  union { float f; unsigned u; } x; x.f = f;
  unsigned r = x.u + 0x7fffu + ((x.u >> 16) & 1u);   // round-to-nearest-even
  unsigned short hv = (unsigned short)(r >> 16);
  bf16 out;
  __builtin_memcpy(&out, &hv, 2);
  return out;
}

static __device__ __forceinline__ v16bf cat16(v8bf lo, v8bf hi) {
  return __builtin_shufflevector(lo, hi, 0,1,2,3,4,5,6,7,8,9,10,11,12,13,14,15);
}

static __device__ __forceinline__ float sigmoidf_(float x) {
  return 1.0f / (1.0f + expf(-x));
}

// ---------------------------------------------------------------------------
// WMMA GEMM: C[M,N](f32) = A[M,KP](bf16) @ Bt[N,KP](bf16)^T  (+ bias[N])
// KP is compile-time (128 or 384). Block = 8 waves, wave = 16 rows x 64 cols.
// B panel (64 x KP) staged in LDS with a 4-bank row skew; per K-step:
//   - next A fragment loaded unconditionally (register-rotated, unroll 2)
//   - all 4 B fragments ds_load'ed as one clause -> single dscnt wait
//   - 4 independent back-to-back v_wmma
// ---------------------------------------------------------------------------
template<int KP>
__global__ void __launch_bounds__(256)
wmma_gemm_bf16(const bf16* __restrict__ A, const bf16* __restrict__ Bt,
               const float* __restrict__ bias, float* __restrict__ C,
               int M, int N)
{
  constexpr int KPP = KP + 8;              // +16B row stride => 4-bank skew
  __shared__ bf16 lds[64 * KPP];

  const int lane  = threadIdx.x & 31;
  const int wave  = threadIdx.x >> 5;
  const int mbase = blockIdx.x * 128 + wave * 16;
  const int nblk  = blockIdx.y * 64;

  // cooperative fill of the B panel (v8bf = 16B chunks, coalesced)
  {
    constexpr int cpr   = KP / 8;          // 16B chunks per row
    constexpr int total = 64 * cpr;
#pragma unroll 1
    for (int idx = threadIdx.x; idx < total; idx += 256) {
      int n  = idx / cpr;
      int ck = idx - n * cpr;
      v8bf v = *reinterpret_cast<const v8bf*>(Bt + (size_t)(nblk + n) * KP + ck * 8);
      *reinterpret_cast<v8bf*>(lds + n * KPP + ck * 8) = v;
    }
  }
  __syncthreads();
  if (mbase >= M) return;                  // wave-uniform; after the only barrier

  const int half   = lane >> 4;            // 0: lanes 0-15, 1: lanes 16-31
  const int l16    = lane & 15;
  const int aoff   = half * 8;             // A: upper half-wave covers K+8/K+24
  const int half16 = half * 8;             // B: upper half-wave covers K+8/K+24

  v8f acc[4];
#pragma unroll
  for (int t = 0; t < 4; ++t) acc[t] = (v8f){0.f,0.f,0.f,0.f,0.f,0.f,0.f,0.f};

  const bf16* arow = A + (size_t)(mbase + l16) * KP;

  // software-pipelined A fragment (over-reads <=64B past last row: harmless,
  // workspace buffers are padded and the loaded values are never consumed)
  v8bf alo = *reinterpret_cast<const v8bf*>(arow + aoff);
  v8bf ahi = *reinterpret_cast<const v8bf*>(arow + 16 + aoff);

#pragma unroll 2
  for (int s = 0; s < KP / 32; ++s) {
    const int kb = s * 32;
    v16bf afrag = cat16(alo, ahi);
    alo = *reinterpret_cast<const v8bf*>(arow + kb + 32 + aoff);
    ahi = *reinterpret_cast<const v8bf*>(arow + kb + 48 + aoff);
    __builtin_prefetch((const void*)(arow + kb + 96), 0, 1);

    v16bf bfr[4];
#pragma unroll
    for (int t = 0; t < 4; ++t) {
      const bf16* brow = lds + (t * 16 + l16) * KPP + kb + half16;
      v8bf blo = *reinterpret_cast<const v8bf*>(brow);
      v8bf bhi = *reinterpret_cast<const v8bf*>(brow + 16);
      bfr[t] = cat16(blo, bhi);
    }
#pragma unroll
    for (int t = 0; t < 4; ++t)
      acc[t] = __builtin_amdgcn_wmma_f32_16x16x32_bf16(
          false, afrag, false, bfr[t], (short)0, acc[t], false, false);
  }

  const int rowoff = half * 8;
#pragma unroll
  for (int t = 0; t < 4; ++t) {
    int ncol = nblk + t * 16 + l16;
    float bb = bias ? bias[ncol] : 0.f;
#pragma unroll
    for (int r = 0; r < 8; ++r) {
      int row = mbase + r + rowoff;        // C layout: vgpr r -> M=r (+8 hi half)
      C[(size_t)row * N + ncol] = acc[t][r] + bb;
    }
  }
}

// ---------------------------------------------------------------------------
// Elementwise / utility kernels
// ---------------------------------------------------------------------------
__global__ void fill_kernel(float* p, float v, size_t n) {
  size_t i = blockIdx.x * (size_t)blockDim.x + threadIdx.x;
  size_t stride = (size_t)gridDim.x * blockDim.x;
  for (; i < n; i += stride) p[i] = v;
}

__global__ void cast_bf16_kernel(const float* __restrict__ in, bf16* __restrict__ out, size_t n) {
  size_t i = blockIdx.x * (size_t)blockDim.x + threadIdx.x;
  size_t stride = (size_t)gridDim.x * blockDim.x;
  for (; i < n; i += stride) out[i] = to_bf16(in[i]);
}

// x [M,K] f32 -> out [M,Kp] bf16 (zero-padded K)
__global__ void cast_pad_kernel(const float* __restrict__ in, bf16* __restrict__ out,
                                int M, int K, int Kp) {
  size_t total = (size_t)M * Kp;
  size_t i = blockIdx.x * (size_t)blockDim.x + threadIdx.x;
  size_t stride = (size_t)gridDim.x * blockDim.x;
  for (; i < total; i += stride) {
    int r = (int)(i / Kp), k = (int)(i % Kp);
    out[i] = (k < K) ? to_bf16(in[(size_t)r * K + k]) : to_bf16(0.f);
  }
}

// w [K,N] f32 -> out [N,Kp] bf16 (B transposed for WMMA, K zero-padded)
__global__ void transpose_bf16_kernel(const float* __restrict__ w, bf16* __restrict__ out,
                                      int K, int N, int Kp) {
  size_t total = (size_t)N * Kp;
  size_t i = blockIdx.x * (size_t)blockDim.x + threadIdx.x;
  size_t stride = (size_t)gridDim.x * blockDim.x;
  for (; i < total; i += stride) {
    int n = (int)(i / Kp), k = (int)(i % Kp);
    out[i] = (k < K) ? to_bf16(w[(size_t)k * N + n]) : to_bf16(0.f);
  }
}

// per-column sum / sumsq with block-chunked atomics
__global__ void colstats_kernel(const float* __restrict__ x, int M, int C,
                                float* __restrict__ sum, float* __restrict__ sumsq,
                                int rows_per_block) {
  int c = blockIdx.x * blockDim.x + threadIdx.x;
  if (c >= C) return;
  int r0 = blockIdx.y * rows_per_block;
  int r1 = r0 + rows_per_block; if (r1 > M) r1 = M;
  float s = 0.f, q = 0.f;
  for (int r = r0; r < r1; ++r) {
    float v = x[(size_t)r * C + c];
    s += v; q += v * v;
  }
  atomicAdd(&sum[c], s);
  atomicAdd(&sumsq[c], q);
}

// BatchNorm apply (+optional skip add, relu, f32 mirror, bf16 mirror)
__global__ void bn_apply_kernel(const float* __restrict__ x,
                                const float* __restrict__ sum,
                                const float* __restrict__ sumsq,
                                const float* __restrict__ g,
                                const float* __restrict__ b,
                                const float* __restrict__ skip,
                                float* __restrict__ out,
                                float* __restrict__ out2,
                                bf16* __restrict__ outb,
                                int M, int C, int relu) {
  size_t total = (size_t)M * C;
  size_t i = blockIdx.x * (size_t)blockDim.x + threadIdx.x;
  size_t stride = (size_t)gridDim.x * blockDim.x;
  float invM = 1.0f / (float)M;
  for (; i < total; i += stride) {
    int c = (int)(i % C);
    float mean = sum[c] * invM;
    float var  = sumsq[c] * invM - mean * mean;
    float v = (x[i] - mean) * rsqrtf(var + BN_EPS) * g[c] + b[c];
    if (skip) v += skip[i];
    if (relu) v = fmaxf(v, 0.f);
    out[i] = v;
    if (out2) out2[i] = v;
    if (outb) outb[i] = to_bf16(v);
  }
}

// scatter-add over edges: agg[dst[e]] += m[src[e]]
__global__ void scatter_add_kernel(const float* __restrict__ m,
                                   const int* __restrict__ src,
                                   const int* __restrict__ dst,
                                   float* __restrict__ agg, int n_edges) {
  int e = blockIdx.x;
  if (e >= n_edges) return;
  int s = src[e], d = dst[e];
  const float* mi = m + (size_t)s * HD;
  float* ao = agg + (size_t)d * HD;
  for (int f = threadIdx.x; f < HD; f += blockDim.x)
    atomicAdd(&ao[f], mi[f]);
}

// fused GRU cell + bf16 mirror of new h
__global__ void gru_kernel(const float* __restrict__ gi, const float* __restrict__ gh,
                           float* __restrict__ h, bf16* __restrict__ hb) {
  size_t total = (size_t)NN * HD;
  size_t i = blockIdx.x * (size_t)blockDim.x + threadIdx.x;
  size_t stride = (size_t)gridDim.x * blockDim.x;
  for (; i < total; i += stride) {
    int n = (int)(i / HD), f = (int)(i % HD);
    const float* gin = gi + (size_t)n * H3;
    const float* ghn = gh + (size_t)n * H3;
    float r  = sigmoidf_(gin[f]          + ghn[f]);
    float z  = sigmoidf_(gin[HD + f]     + ghn[HD + f]);
    float nn = tanhf    (gin[2 * HD + f] + r * ghn[2 * HD + f]);
    float hv = (1.f - z) * nn + z * h[i];
    h[i] = hv;
    hb[i] = to_bf16(hv);
  }
}

// attention logits: one wave per node, wave32 shuffle reduction
__global__ void att_kernel(const float* __restrict__ h, const float* __restrict__ aw,
                           const float* __restrict__ ab, float* __restrict__ logits) {
  int node = (int)((blockIdx.x * (size_t)blockDim.x + threadIdx.x) >> 5);
  int lane = threadIdx.x & 31;
  if (node >= NN) return;
  const float* hr = h + (size_t)node * HD;
  float s = 0.f;
  for (int k = lane; k < HD; k += 32) s += hr[k] * aw[k];
  for (int off = 16; off > 0; off >>= 1) s += __shfl_down(s, off, 32);
  if (lane == 0) logits[node] = s + ab[0];
}

static __device__ __forceinline__ void atomicMaxFloat(float* addr, float val) {
  unsigned* ai = (unsigned*)addr;
  unsigned old = *ai;
  while (true) {
    float f; __builtin_memcpy(&f, &old, 4);
    if (f >= val) break;
    unsigned nv; __builtin_memcpy(&nv, &val, 4);
    unsigned assumed = old;
    old = atomicCAS(ai, assumed, nv);
    if (old == assumed) break;
  }
}

__global__ void segmax_kernel(const float* __restrict__ logits, const int* __restrict__ batch,
                              float* __restrict__ segmax) {
  size_t i = blockIdx.x * (size_t)blockDim.x + threadIdx.x;
  size_t stride = (size_t)gridDim.x * blockDim.x;
  for (; i < NN; i += stride) atomicMaxFloat(&segmax[batch[i]], logits[i]);
}

__global__ void segexp_kernel(const float* __restrict__ logits, const int* __restrict__ batch,
                              const float* __restrict__ segmax, float* __restrict__ e,
                              float* __restrict__ denom) {
  size_t i = blockIdx.x * (size_t)blockDim.x + threadIdx.x;
  size_t stride = (size_t)gridDim.x * blockDim.x;
  for (; i < NN; i += stride) {
    float v = expf(logits[i] - segmax[batch[i]]);
    e[i] = v;
    atomicAdd(&denom[batch[i]], v);
  }
}

__global__ void pool_kernel(const float* __restrict__ h, const float* __restrict__ e,
                            const float* __restrict__ denom, const int* __restrict__ batch,
                            float* __restrict__ pooled) {
  size_t total = (size_t)NN * HD;
  size_t i = blockIdx.x * (size_t)blockDim.x + threadIdx.x;
  size_t stride = (size_t)gridDim.x * blockDim.x;
  for (; i < total; i += stride) {
    int n = (int)(i / HD), f = (int)(i % HD);
    int g = batch[n];
    float w = e[n] / denom[g];
    atomicAdd(&pooled[(size_t)g * HD + f], w * h[i]);
  }
}

__global__ void fc3_kernel(const float* __restrict__ y, const float* __restrict__ w,
                           const float* __restrict__ b, float* __restrict__ out) {
  int i = blockIdx.x * blockDim.x + threadIdx.x;
  if (i >= NG * 2) return;
  int g = i >> 1, o = i & 1;
  float s = b[o];
  const float* yr = y + (size_t)g * HH;
  for (int k = 0; k < HH; ++k) s += yr[k] * w[k * 2 + o];
  out[i] = s;
}

// ---------------------------------------------------------------------------
extern "C" void kernel_launch(void* const* d_in, const int* in_sizes, int n_in,
                              void* d_out, int out_size, void* d_ws, size_t ws_size,
                              hipStream_t stream) {
  const float* x       = (const float*)d_in[0];
  const int*   eidx    = (const int*)  d_in[1];
  const int*   batch   = (const int*)  d_in[2];
  const float* w_in    = (const float*)d_in[3];
  const float* b_in    = (const float*)d_in[4];
  const float* bn1_g   = (const float*)d_in[5];
  const float* bn1_b   = (const float*)d_in[6];
  const float* ggc_w   = (const float*)d_in[7];
  const float* gru_wih = (const float*)d_in[8];
  const float* gru_whh = (const float*)d_in[9];
  const float* gru_bih = (const float*)d_in[10];
  const float* gru_bhh = (const float*)d_in[11];
  const float* bn2_g   = (const float*)d_in[12];
  const float* bn2_b   = (const float*)d_in[13];
  const float* att_w   = (const float*)d_in[14];
  const float* att_b   = (const float*)d_in[15];
  const float* fc1_w   = (const float*)d_in[16];
  const float* fc1_b   = (const float*)d_in[17];
  const float* bn3_g   = (const float*)d_in[18];
  const float* bn3_b   = (const float*)d_in[19];
  const float* fc2_w   = (const float*)d_in[20];
  const float* fc2_b   = (const float*)d_in[21];
  const float* bn4_g   = (const float*)d_in[22];
  const float* bn4_b   = (const float*)d_in[23];
  const float* fc3_w   = (const float*)d_in[24];
  const float* fc3_b   = (const float*)d_in[25];
  const int* e_src = eidx;
  const int* e_dst = eidx + NE;

  // --- workspace carve-up ---
  char* wp = (char*)d_ws;
  auto alloc = [&](size_t bytes) -> char* {
    char* p = wp; wp += (bytes + 511) & ~(size_t)511; return p;   // pad: GEMM A over-read
  };
  float* f_m     = (float*)alloc(sizeof(float) * (size_t)NN * HD);  // x@w_in out, then msg buf
  float* f_h     = (float*)alloc(sizeof(float) * (size_t)NN * HD);
  float* f_skip  = (float*)alloc(sizeof(float) * (size_t)NN * HD);
  float* f_agg   = (float*)alloc(sizeof(float) * (size_t)NN * HD);
  float* f_gi    = (float*)alloc(sizeof(float) * (size_t)NN * H3);
  float* f_gh    = (float*)alloc(sizeof(float) * (size_t)NN * H3);
  float* f_log   = (float*)alloc(sizeof(float) * NN);
  float* f_e     = (float*)alloc(sizeof(float) * NN);
  float* f_segmx = (float*)alloc(sizeof(float) * NG);
  float* f_denom = (float*)alloc(sizeof(float) * NG);
  float* f_pool  = (float*)alloc(sizeof(float) * (size_t)NG * HD);
  float* f_y1    = (float*)alloc(sizeof(float) * (size_t)NG * HD);
  float* f_y2    = (float*)alloc(sizeof(float) * (size_t)NG * HH);
  float* f_stats = (float*)alloc(sizeof(float) * 2 * HD);           // sum | sumsq
  bf16* b_x    = (bf16*)alloc(sizeof(bf16) * (size_t)NN * KPIN);
  bf16* b_h    = (bf16*)alloc(sizeof(bf16) * (size_t)NN * HD);
  bf16* b_agg  = (bf16*)alloc(sizeof(bf16) * (size_t)NN * HD);
  bf16* b_pool = (bf16*)alloc(sizeof(bf16) * (size_t)NG * HD);
  bf16* b_y1   = (bf16*)alloc(sizeof(bf16) * (size_t)NG * HD);
  bf16* b_win  = (bf16*)alloc(sizeof(bf16) * (size_t)HD * KPIN);
  bf16* b_ggc  = (bf16*)alloc(sizeof(bf16) * (size_t)GSTEPS * HD * HD);
  bf16* b_wih  = (bf16*)alloc(sizeof(bf16) * (size_t)H3 * HD);
  bf16* b_whh  = (bf16*)alloc(sizeof(bf16) * (size_t)H3 * HD);
  bf16* b_fc1  = (bf16*)alloc(sizeof(bf16) * (size_t)HD * HD);
  bf16* b_fc2  = (bf16*)alloc(sizeof(bf16) * (size_t)HH * HD);

  const int T = 256;
  dim3 gsBig(2048), gsMid(512);

  // --- weight prep (transpose to [N,Kp] bf16) ---
  transpose_bf16_kernel<<<gsMid, T, 0, stream>>>(w_in, b_win, DIN, HD, KPIN);
  for (int i = 0; i < GSTEPS; ++i)
    transpose_bf16_kernel<<<gsMid, T, 0, stream>>>(ggc_w + (size_t)i * HD * HD,
                                                   b_ggc + (size_t)i * HD * HD, HD, HD, HD);
  cast_bf16_kernel<<<gsMid, T, 0, stream>>>(gru_wih, b_wih, (size_t)H3 * HD); // already [N,K]
  cast_bf16_kernel<<<gsMid, T, 0, stream>>>(gru_whh, b_whh, (size_t)H3 * HD);
  transpose_bf16_kernel<<<gsMid, T, 0, stream>>>(fc1_w, b_fc1, HD, HD, HD);
  transpose_bf16_kernel<<<gsMid, T, 0, stream>>>(fc2_w, b_fc2, HD, HH, HD);

  // --- input projection: f_m = x @ w_in + b_in ---
  cast_pad_kernel<<<gsBig, T, 0, stream>>>(x, b_x, NN, DIN, KPIN);
  dim3 gNodeH((NN + 127) / 128, HD / 64);
  wmma_gemm_bf16<KPIN><<<gNodeH, T, 0, stream>>>(b_x, b_win, b_in, f_m, NN, HD);

  // --- BN1 + relu -> h, skip, b_h ---
  fill_kernel<<<1, T, 0, stream>>>(f_stats, 0.f, 2 * HD);
  colstats_kernel<<<dim3((HD + 127) / 128, 100), 128, 0, stream>>>(f_m, NN, HD,
                                                                   f_stats, f_stats + HD, 500);
  bn_apply_kernel<<<gsBig, T, 0, stream>>>(f_m, f_stats, f_stats + HD, bn1_g, bn1_b,
                                           nullptr, f_h, f_skip, b_h, NN, HD, 1);

  // --- GatedGraphConv steps ---
  dim3 gNode3H((NN + 127) / 128, H3 / 64);
  for (int i = 0; i < GSTEPS; ++i) {
    wmma_gemm_bf16<HD><<<gNodeH, T, 0, stream>>>(b_h, b_ggc + (size_t)i * HD * HD,
                                                 nullptr, f_m, NN, HD);
    fill_kernel<<<gsBig, T, 0, stream>>>(f_agg, 0.f, (size_t)NN * HD);
    scatter_add_kernel<<<NE, 128, 0, stream>>>(f_m, e_src, e_dst, f_agg, NE);
    cast_bf16_kernel<<<gsBig, T, 0, stream>>>(f_agg, b_agg, (size_t)NN * HD);
    wmma_gemm_bf16<HD><<<gNode3H, T, 0, stream>>>(b_agg, b_wih, gru_bih, f_gi, NN, H3);
    wmma_gemm_bf16<HD><<<gNode3H, T, 0, stream>>>(b_h,   b_whh, gru_bhh, f_gh, NN, H3);
    gru_kernel<<<gsBig, T, 0, stream>>>(f_gi, f_gh, f_h, b_h);
  }

  // --- BN2 + skip + relu (in place on h) ---
  fill_kernel<<<1, T, 0, stream>>>(f_stats, 0.f, 2 * HD);
  colstats_kernel<<<dim3((HD + 127) / 128, 100), 128, 0, stream>>>(f_h, NN, HD,
                                                                   f_stats, f_stats + HD, 500);
  bn_apply_kernel<<<gsBig, T, 0, stream>>>(f_h, f_stats, f_stats + HD, bn2_g, bn2_b,
                                           f_skip, f_h, nullptr, nullptr, NN, HD, 1);

  // --- attention pooling ---
  att_kernel<<<(NN * 32 + T - 1) / T, T, 0, stream>>>(f_h, att_w, att_b, f_log);
  fill_kernel<<<1, T, 0, stream>>>(f_segmx, -3.0e38f, NG);
  fill_kernel<<<1, T, 0, stream>>>(f_denom, 0.f, NG);
  fill_kernel<<<gsMid, T, 0, stream>>>(f_pool, 0.f, (size_t)NG * HD);
  segmax_kernel<<<gsMid, T, 0, stream>>>(f_log, batch, f_segmx);
  segexp_kernel<<<gsMid, T, 0, stream>>>(f_log, batch, f_segmx, f_e, f_denom);
  pool_kernel<<<gsBig, T, 0, stream>>>(f_h, f_e, f_denom, batch, f_pool);

  // --- head: fc1 -> bn3 -> relu -> fc2 -> bn4 -> relu -> fc3 ---
  cast_bf16_kernel<<<gsMid, T, 0, stream>>>(f_pool, b_pool, (size_t)NG * HD);
  dim3 gHead1((NG + 127) / 128, HD / 64);
  wmma_gemm_bf16<HD><<<gHead1, T, 0, stream>>>(b_pool, b_fc1, fc1_b, f_y1, NG, HD);
  fill_kernel<<<1, T, 0, stream>>>(f_stats, 0.f, 2 * HD);
  colstats_kernel<<<dim3((HD + 127) / 128, 1), 128, 0, stream>>>(f_y1, NG, HD,
                                                                 f_stats, f_stats + HD, NG);
  bn_apply_kernel<<<gsMid, T, 0, stream>>>(f_y1, f_stats, f_stats + HD, bn3_g, bn3_b,
                                           nullptr, f_y1, nullptr, b_y1, NG, HD, 1);
  dim3 gHead2((NG + 127) / 128, HH / 64);
  wmma_gemm_bf16<HD><<<gHead2, T, 0, stream>>>(b_y1, b_fc2, fc2_b, f_y2, NG, HH);
  fill_kernel<<<1, T, 0, stream>>>(f_stats, 0.f, 2 * HD);
  colstats_kernel<<<dim3((HH + 127) / 128, 1), 128, 0, stream>>>(f_y2, NG, HH,
                                                                 f_stats, f_stats + HD, NG);
  bn_apply_kernel<<<gsMid, T, 0, stream>>>(f_y2, f_stats, f_stats + HD, bn4_g, bn4_b,
                                           nullptr, f_y2, nullptr, nullptr, NG, HH, 1);
  fc3_kernel<<<(NG * 2 + T - 1) / T, T, 0, stream>>>(f_y2, fc3_w, fc3_b, (float*)d_out);
}
